// DeltaNet_31877247271509
// MI455X (gfx1250) — matrix operand order repaired and checked by reading, not compiled, not run on hardware
//
#include <hip/hip_runtime.h>
#include <hip/hip_bf16.h>
#include <math.h>

// ---------------------------------------------------------------------------
// Types for CDNA5 WMMA (wave32, 16x16x32 bf16, f32 accumulate)
// ---------------------------------------------------------------------------
typedef __bf16 bf16;
typedef bf16  v16bf __attribute__((ext_vector_type(16)));
typedef float v8f   __attribute__((ext_vector_type(8)));
typedef unsigned int u32x4 __attribute__((ext_vector_type(4)));
typedef int i32x4 __attribute__((ext_vector_type(4)));
typedef int i32x8 __attribute__((ext_vector_type(8)));

#define DEVINL __device__ __forceinline__

#if defined(__has_builtin)
#  if __has_builtin(__builtin_amdgcn_tensor_load_to_lds)
#    define HAVE_TDM 1
#  endif
#  if __has_builtin(__builtin_amdgcn_s_wait_tensorcnt)
#    define HAVE_WAIT_TCNT 1
#  endif
#endif

DEVINL v8f wmma_bf16(v16bf a, v16bf b, v8f c) {
  // (neg_a, A, neg_b, B, c_mod, C, reuse_a, reuse_b)
  return __builtin_amdgcn_wmma_f32_16x16x32_bf16(false, a, false, b, (short)0, c,
                                                 false, false);
}

DEVINL void wait_tensorcnt0() {
#ifdef HAVE_WAIT_TCNT
  __builtin_amdgcn_s_wait_tensorcnt(0);
#else
  asm volatile("s_wait_tensorcnt 0x0" ::: "memory");
#endif
}

// ---------------------------------------------------------------------------
// TDM: DMA a [rows x cols] bf16 tile (row stride = ld elements) from global
// into contiguous LDS at lds_addr. Issued by one wave; tracked by TENSORcnt.
// D# layout per CDNA5 ISA ch.8 (group0: count/lds/global/type, group1: dims).
// ---------------------------------------------------------------------------
DEVINL void tdm_load_tile(const bf16* gsrc, unsigned lds_addr, int rows, int cols,
                          int ld) {
#ifdef HAVE_TDM
  unsigned long long ga = (unsigned long long)(size_t)gsrc;
  u32x4 g0;
  g0[0] = 1u;                                        // count=1 valid user D#
  g0[1] = lds_addr;                                  // LDS byte address
  g0[2] = (unsigned)(ga & 0xFFFFFFFFull);            // global_addr[31:0]
  g0[3] = (unsigned)((ga >> 32) & 0x01FFFFFFull) | 0x80000000u;  // [56:32]|type=2
  i32x8 g1;
  g1[0] = 1 << 16;                                   // data_size=1 (2B elements)
  g1[1] = (int)(((unsigned)ld & 0xFFFFu) << 16);     // tensor_dim0[15:0]
  g1[2] = (int)((((unsigned)rows & 0xFFFFu) << 16) | (((unsigned)ld >> 16) & 0xFFFFu));
  g1[3] = (int)(((unsigned)cols & 0xFFFFu) << 16);   // tile_dim0 = cols
  g1[4] = (int)((unsigned)rows & 0xFFFFu);           // tile_dim1 = rows
  g1[5] = ld;                                        // tensor_dim0_stride[31:0]
  g1[6] = 0;
  g1[7] = 0;
  i32x4 gz = {0, 0, 0, 0};
#if defined(__clang_major__) && (__clang_major__ >= 23)
  i32x8 gz8 = {0, 0, 0, 0, 0, 0, 0, 0};
  __builtin_amdgcn_tensor_load_to_lds(g0, g1, gz, gz, gz8, 0);
#else
  __builtin_amdgcn_tensor_load_to_lds(g0, g1, gz, gz, 0);
#endif
#else
  (void)gsrc; (void)lds_addr; (void)rows; (void)cols; (void)ld;
#endif
}

// Stage a 64x32 bf16 B tile into LDS (TDM if available, else cooperative copy)
DEVINL void stage_b(const bf16* B, int ldb, int n0, int k0, bf16* dst, int tid) {
#ifdef HAVE_TDM
  if (tid < 32)
    tdm_load_tile(B + (size_t)n0 * ldb + k0, (unsigned)(size_t)dst, 64, 32, ldb);
#else
  for (int i = tid; i < 64 * 32; i += 128) {
    int r = i >> 5, c = i & 31;
    dst[i] = B[(size_t)(n0 + r) * ldb + k0 + c];
  }
#endif
}
DEVINL void stage_b_wait(int tid) {
#ifdef HAVE_TDM
  if (tid < 32) wait_tensorcnt0();
#else
  (void)tid;
#endif
}

// A fragment, A stored row-major [M,K] (K contiguous).
// A 16x32 layout (matches B): lane<16 holds M=lane, K in k0..k0+15;
// lane>=16 holds M=lane-16, K in k0+16..k0+31.
DEVINL v16bf load_a_rm(const bf16* A, int lda, int m0, int k0, int lane) {
  int m  = m0 + (lane & 15);
  int kb = k0 + ((lane & 16) ? 16 : 0);
  const bf16* p = A + (size_t)m * lda + kb;
  v16bf f;
#pragma unroll
  for (int j = 0; j < 16; ++j) f[j] = p[j];
  return f;
}

// A fragment where logical A[m][k] is stored k-major: addr = k*ldm + m  (e.g. k^T)
DEVINL v16bf load_a_km(const bf16* A, int ldm, int m0, int k0, int lane) {
  int m  = m0 + (lane & 15);
  int kb = k0 + ((lane & 16) ? 16 : 0);
  v16bf f;
#pragma unroll
  for (int j = 0; j < 16; ++j) f[j] = A[(size_t)(kb + j) * ldm + m];
  return f;
}

// B fragment, logical B[k][n] stored as [N,K] row-major (i.e. weight W[n,k]).
// B 32x16 layout used here (contiguous k, same as A): lanes 0-15 hold K=k0..k0+15 for N=lane&15, lanes 16-31 K=k0+16..31.
DEVINL v16bf load_b_nk(const bf16* B, int ldb, int n0, int k0, int lane) {
  int n  = n0 + (lane & 15);
  int kb = k0 + ((lane & 16) ? 16 : 0);
  const bf16* p = B + (size_t)n * ldb + kb;
  v16bf f;
#pragma unroll
  for (int j = 0; j < 16; ++j) f[j] = p[j];
  return f;
}

// B fragment, logical B[k][n] stored as [K,N] row-major (N contiguous).
DEVINL v16bf load_b_kn(const bf16* B, int ldn, int n0, int k0, int lane) {
  int n  = n0 + (lane & 15);
  int kb = k0 + ((lane & 16) ? 16 : 0);
  v16bf f;
#pragma unroll
  for (int j = 0; j < 16; ++j) f[j] = B[(size_t)(kb + j) * ldn + n];
  return f;
}

// C/D fragment (f32, 8 VGPR): lane holds N=n0+(lane&15), rows m0+r+8*(lane>=16)
DEVINL v8f load_c_bf16(const bf16* C, int ldc, int m0, int n0, int lane) {
  int n  = n0 + (lane & 15);
  int mb = m0 + ((lane & 16) ? 8 : 0);
  v8f c;
#pragma unroll
  for (int r = 0; r < 8; ++r) c[r] = (float)C[(size_t)(mb + r) * ldc + n];
  return c;
}
DEVINL void store_c_bf16(bf16* C, int ldc, int m0, int n0, int lane, v8f c) {
  int n  = n0 + (lane & 15);
  int mb = m0 + ((lane & 16) ? 8 : 0);
#pragma unroll
  for (int r = 0; r < 8; ++r) C[(size_t)(mb + r) * ldc + n] = (bf16)c[r];
}
DEVINL void store_c_f32(float* C, size_t ldc, int m0, int n0, int lane, v8f c) {
  int n  = n0 + (lane & 15);
  int mb = m0 + ((lane & 16) ? 8 : 0);
#pragma unroll
  for (int r = 0; r < 8; ++r) C[(size_t)(mb + r) * ldc + n] = c[r];
}

// ---------------------------------------------------------------------------
// WMMA GEMM: C[M,N] = A[M,K] * B[N,K]^T   (A,B bf16, C f32).
// Block = 128 threads = 4 waves, 64x64 block tile, 16x64 per wave.
// B tiles staged to LDS by the Tensor Data Mover, double buffered so the
// next K-slice DMA overlaps the current slice's WMMAs.
// ---------------------------------------------------------------------------
__global__ __launch_bounds__(128) void gemm_bf16(const bf16* __restrict__ A,
                                                 const bf16* __restrict__ B,
                                                 float* __restrict__ C,
                                                 int M, int N, int K) {
  __shared__ bf16 bst[2][64 * 32];
  int tid = threadIdx.x;
  int lane = tid & 31;
  int wave = tid >> 5;
  int m0 = blockIdx.x * 64 + wave * 16;
  int n0 = blockIdx.y * 64;
  v8f acc0 = {}, acc1 = {}, acc2 = {}, acc3 = {};
  // prologue: stage first B tile
  stage_b(B, K, n0, 0, &bst[0][0], tid);
  stage_b_wait(tid);
  __syncthreads();
  for (int k0 = 0; k0 < K; k0 += 32) {
    int buf = (k0 >> 5) & 1;
    if (k0 + 32 < K) stage_b(B, K, n0, k0 + 32, &bst[buf ^ 1][0], tid);
    __builtin_prefetch(A + (size_t)(m0 + (lane & 15)) * K + k0 + 256, 0, 3);
    v16bf a = load_a_rm(A, K, m0, k0, lane);
    const bf16* bl = &bst[buf][0];
    acc0 = wmma_bf16(a, load_b_nk(bl, 32, 0, 0, lane), acc0);
    acc1 = wmma_bf16(a, load_b_nk(bl, 32, 16, 0, lane), acc1);
    acc2 = wmma_bf16(a, load_b_nk(bl, 32, 32, 0, lane), acc2);
    acc3 = wmma_bf16(a, load_b_nk(bl, 32, 48, 0, lane), acc3);
    if (k0 + 32 < K) stage_b_wait(tid);
    __syncthreads();
  }
  store_c_f32(C, (size_t)N, m0, n0, lane, acc0);
  store_c_f32(C, (size_t)N, m0, n0 + 16, lane, acc1);
  store_c_f32(C, (size_t)N, m0, n0 + 32, lane, acc2);
  store_c_f32(C, (size_t)N, m0, n0 + 48, lane, acc3);
}

// ---------------------------------------------------------------------------
// Elementwise helpers
// ---------------------------------------------------------------------------
__global__ void f32_to_bf16_k(const float* __restrict__ src, bf16* __restrict__ dst,
                              int n) {
  int i = blockIdx.x * blockDim.x + threadIdx.x;
  if (i < n) dst[i] = (bf16)src[i];
}

__global__ void pad_g1_k(const float* __restrict__ src, bf16* __restrict__ dst) {
  int i = blockIdx.x * blockDim.x + threadIdx.x;  // 2048*1088
  if (i >= 2048 * 1088) return;
  int n = i / 1088, k = i % 1088;
  dst[i] = (k < 1084) ? (bf16)src[(size_t)n * 1084 + k] : (bf16)0.0f;
}

__global__ void pad_g2_k(const float* __restrict__ src, bf16* __restrict__ dst) {
  int i = blockIdx.x * blockDim.x + threadIdx.x;  // 64*2048
  if (i >= 64 * 2048) return;
  int n = i / 2048, k = i % 2048;
  dst[i] = (n < 20) ? (bf16)src[(size_t)n * 2048 + k] : (bf16)0.0f;
}

// Depthwise causal FIR along L: y[b,l,c] = sum_t w[c,t] * x[b, l-(K-1)+t, c]
__global__ void dwconv_causal_k(const float* __restrict__ x, const float* __restrict__ w,
                                float* __restrict__ y, int L, int C, int K,
                                int act_silu, long total) {
  long i = (long)blockIdx.x * blockDim.x + threadIdx.x;
  if (i >= total) return;
  int c = (int)(i % C);
  int l = (int)((i / C) % L);
  long bL = (i / C) - l;  // b*L
  float acc = 0.0f;
  for (int t = 0; t < K; ++t) {
    int ls = l - (K - 1) + t;
    if (ls >= 0) acc += w[(size_t)c * K + t] * x[(bL + ls) * (size_t)C + c];
  }
  if (act_silu) acc = acc / (1.0f + expf(-acc));
  y[i] = acc;
}

__global__ void beta_k(const float* __restrict__ h, const float* __restrict__ bw,
                       float* __restrict__ beta) {
  int i = blockIdx.x * blockDim.x + threadIdx.x;  // 4096*4
  if (i >= 4096 * 4) return;
  int hh = i & 3, row = i >> 2;
  const float* hp = h + (size_t)row * 1024;
  const float* wp = bw + (size_t)hh * 1024;
  float s = 0.0f;
  for (int k = 0; k < 1024; ++k) s += hp[k] * wp[k];
  beta[i] = 1.0f / (1.0f + expf(-s));
}

__global__ void gelu_bias_cvt_k(const float* __restrict__ x, const float* __restrict__ bias,
                                bf16* __restrict__ y, int N, long total) {
  long i = (long)blockIdx.x * blockDim.x + threadIdx.x;
  if (i >= total) return;
  float v = x[i] + bias[i % N];
  y[i] = (bf16)(0.5f * v * (1.0f + erff(v * 0.70710678118654752f)));
}

// ---------------------------------------------------------------------------
// Chunked delta rule: one block per (b,h). 8 waves, S[256x256] bf16 in LDS.
// All matmuls via v_wmma_f32_16x16x32_bf16.
// ---------------------------------------------------------------------------
__global__ __launch_bounds__(256) void delta_kernel(const float* __restrict__ qc,
                                                    const float* __restrict__ kc,
                                                    const float* __restrict__ vc,
                                                    const float* __restrict__ beta,
                                                    float* __restrict__ dout) {
  __shared__ bf16 S[256 * 256];      // 128 KB recurrent state (dk x dv)
  __shared__ bf16 qT[32 * 256];      // chunk tiles, token-major
  __shared__ bf16 kT[32 * 256];
  __shared__ bf16 kbT[32 * 256];
  __shared__ bf16 vbT[32 * 256];
  __shared__ bf16 uM[32 * 256];
  __shared__ bf16 wM[32 * 256];      // holds -(inv @ k_beta)
  __shared__ bf16 uadj[32 * 256];
  __shared__ bf16 invM[32 * 32];
  __shared__ bf16 attnM[32 * 32];
  __shared__ float Af[32 * 32];

  int tid = threadIdx.x, lane = tid & 31, wave = tid >> 5;
  int b = blockIdx.x >> 2, h = blockIdx.x & 3;
  const float* qbase = qc + (size_t)b * 2048 * 1024 + h * 256;
  const float* kbase = kc + (size_t)b * 2048 * 1024 + h * 256;
  const float* vbase = vc + (size_t)b * 2048 * 1024 + h * 256;
  const float* betab = beta + (size_t)b * 2048 * 4 + h;

  for (int i = tid; i < 256 * 256; i += 256) S[i] = (bf16)0.0f;
  __syncthreads();

  for (int ch = 0; ch < 64; ++ch) {
    int l0 = ch * 32;
    // ---- P1: load chunk, l2-normalize q,k; v*beta; k*beta -> bf16 LDS ----
#pragma unroll
    for (int tt = 0; tt < 4; ++tt) {
      int t = wave * 4 + tt;
      const float* qp = qbase + (size_t)(l0 + t) * 1024 + lane * 8;
      const float* kp = kbase + (size_t)(l0 + t) * 1024 + lane * 8;
      const float* vp = vbase + (size_t)(l0 + t) * 1024 + lane * 8;
      float qv[8], kv[8], vv[8], sq = 0.0f, sk = 0.0f;
#pragma unroll
      for (int j = 0; j < 8; ++j) {
        qv[j] = qp[j]; sq += qv[j] * qv[j];
        kv[j] = kp[j]; sk += kv[j] * kv[j];
        vv[j] = vp[j];
      }
#pragma unroll
      for (int m = 16; m >= 1; m >>= 1) {
        sq += __shfl_xor(sq, m, 32);
        sk += __shfl_xor(sk, m, 32);
      }
      float qn = rsqrtf(sq + 1e-6f), kn = rsqrtf(sk + 1e-6f);
      float bt = betab[(size_t)(l0 + t) * 4];
#pragma unroll
      for (int j = 0; j < 8; ++j) {
        int d = lane * 8 + j;
        float kq = kv[j] * kn;
        qT[t * 256 + d]  = (bf16)(qv[j] * qn);
        kT[t * 256 + d]  = (bf16)kq;
        kbT[t * 256 + d] = (bf16)(kq * bt);
        vbT[t * 256 + d] = (bf16)(vv[j] * bt);
      }
    }
    __syncthreads();
    // ---- P2: waves 0-3: A = kb @ k^T (f32); waves 4-7: attn = tril(q @ k^T) ----
    {
      int mi = (wave & 3) >> 1, ni = wave & 1;
      const bf16* amat = (wave < 4) ? kbT : qT;
      v8f acc = {};
      for (int k0 = 0; k0 < 256; k0 += 32) {
        v16bf a  = load_a_rm(amat, 256, mi * 16, k0, lane);
        v16bf bb = load_b_nk(kT, 256, ni * 16, k0, lane);
        acc = wmma_bf16(a, bb, acc);
      }
      int n = ni * 16 + (lane & 15);
      int mb = mi * 16 + ((lane & 16) ? 8 : 0);
      if (wave < 4) {
#pragma unroll
        for (int r = 0; r < 8; ++r) Af[(mb + r) * 32 + n] = acc[r];
      } else {
#pragma unroll
        for (int r = 0; r < 8; ++r)
          attnM[(mb + r) * 32 + n] = (bf16)((mb + r >= n) ? acc[r] : 0.0f);
      }
    }
    __syncthreads();
    // ---- P3: inv = (I + strict_tril(A))^-1 by forward substitution (wave 0) ----
    if (wave == 0) {
      float col[32];
#pragma unroll
      for (int i = 0; i < 32; ++i) col[i] = (i == lane) ? 1.0f : 0.0f;
#pragma unroll
      for (int i = 1; i < 32; ++i) {
        float s = 0.0f;
#pragma unroll
        for (int j = 0; j < i; ++j) s += Af[i * 32 + j] * col[j];
        col[i] -= s;
      }
#pragma unroll
      for (int i = 0; i < 32; ++i) invM[i * 32 + lane] = (bf16)col[i];
    }
    __syncthreads();
    // ---- P4: u = inv @ v_beta ; wM = -(inv @ k_beta). 64 tile-jobs / 8 waves ----
#pragma unroll
    for (int t = 0; t < 8; ++t) {
      int jt = wave * 8 + t;
      int mat = jt >> 5, rem = jt & 31, mi = rem >> 4, ni = rem & 15;
      v16bf a  = load_a_rm(invM, 32, mi * 16, 0, lane);
      v16bf bb = load_b_kn(mat ? kbT : vbT, 256, ni * 16, 0, lane);
      v8f acc = {};
      acc = wmma_bf16(a, bb, acc);
      if (mat) {
#pragma unroll
        for (int r = 0; r < 8; ++r) acc[r] = -acc[r];  // store negated w
        store_c_bf16(wM, 256, mi * 16, ni * 16, lane, acc);
      } else {
        store_c_bf16(uM, 256, mi * 16, ni * 16, lane, acc);
      }
    }
    __syncthreads();
    // ---- P6: u_adj = u + (-w) @ S ----
#pragma unroll
    for (int t = 0; t < 4; ++t) {
      int jt = wave * 4 + t, mi = jt >> 4, ni = jt & 15;
      v8f acc = load_c_bf16(uM, 256, mi * 16, ni * 16, lane);
      for (int k0 = 0; k0 < 256; k0 += 32) {
        v16bf a  = load_a_rm(wM, 256, mi * 16, k0, lane);
        v16bf bb = load_b_kn(S, 256, ni * 16, k0, lane);
        acc = wmma_bf16(a, bb, acc);
      }
      store_c_bf16(uadj, 256, mi * 16, ni * 16, lane, acc);
    }
    __syncthreads();
    // ---- P7: o = q @ S + attn @ u_adj -> global ----
#pragma unroll
    for (int t = 0; t < 4; ++t) {
      int jt = wave * 4 + t, mi = jt >> 4, ni = jt & 15;
      v8f acc = {};
      for (int k0 = 0; k0 < 256; k0 += 32) {
        v16bf a  = load_a_rm(qT, 256, mi * 16, k0, lane);
        v16bf bb = load_b_kn(S, 256, ni * 16, k0, lane);
        acc = wmma_bf16(a, bb, acc);
      }
      {
        v16bf a  = load_a_rm(attnM, 32, mi * 16, 0, lane);
        v16bf bb = load_b_kn(uadj, 256, ni * 16, 0, lane);
        acc = wmma_bf16(a, bb, acc);
      }
      int n = ni * 16 + (lane & 15);
      int mb = mi * 16 + ((lane & 16) ? 8 : 0);
#pragma unroll
      for (int r = 0; r < 8; ++r)
        dout[((size_t)b * 2048 + l0 + mb + r) * 1024 + h * 256 + n] = acc[r];
    }
    __syncthreads();
    // ---- P8: S += k^T @ u_adj (256 tiles / 8 waves) ----
    for (int t = 0; t < 32; ++t) {
      int jt = wave * 32 + t, mi = jt >> 4, ni = jt & 15;
      v16bf a  = load_a_km(kT, 256, mi * 16, 0, lane);
      v16bf bb = load_b_kn(uadj, 256, ni * 16, 0, lane);
      v8f c = load_c_bf16(S, 256, mi * 16, ni * 16, lane);
      c = wmma_bf16(a, bb, c);
      store_c_bf16(S, 256, mi * 16, ni * 16, lane, c);
    }
    __syncthreads();
  }
}

// ---------------------------------------------------------------------------
// Stats + gate-input assembly: gin[row, 0:1024]=hidden(bf16), then per source
// (short,long,wide,delta,v): mean[4], rms[4], max_abs[4]; cols 1084..1087 = 0.
// ---------------------------------------------------------------------------
__global__ __launch_bounds__(128) void stats_kernel(
    const float* __restrict__ fshort, const float* __restrict__ flong,
    const float* __restrict__ fwide, const float* __restrict__ ddelta,
    const float* __restrict__ vv, const bf16* __restrict__ hbf,
    bf16* __restrict__ gin) {
  int row = blockIdx.x;
  int tid = threadIdx.x, lane = tid & 31, h = tid >> 5;
  bf16* out = gin + (size_t)row * 1088;
#pragma unroll
  for (int j = 0; j < 8; ++j) out[tid * 8 + j] = hbf[(size_t)row * 1024 + tid * 8 + j];
  if (tid < 4) out[1084 + tid] = (bf16)0.0f;
  const float* srcs[5] = {fshort, flong, fwide, ddelta, vv};
#pragma unroll
  for (int s = 0; s < 5; ++s) {
    const float* p = srcs[s] + (size_t)row * 1024 + h * 256 + lane * 8;
    float sum = 0.0f, sq = 0.0f, mx = 0.0f;
#pragma unroll
    for (int j = 0; j < 8; ++j) {
      float x = p[j];
      sum += x; sq += x * x; mx = fmaxf(mx, fabsf(x));
    }
#pragma unroll
    for (int m = 16; m >= 1; m >>= 1) {
      sum += __shfl_xor(sum, m, 32);
      sq += __shfl_xor(sq, m, 32);
      mx = fmaxf(mx, __shfl_xor(mx, m, 32));
    }
    if (lane == 0) {
      out[1024 + s * 12 + 0 + h] = (bf16)(sum * (1.0f / 256.0f));
      out[1024 + s * 12 + 4 + h] = (bf16)sqrtf(fmaxf(sq * (1.0f / 256.0f), 1e-8f));
      out[1024 + s * 12 + 8 + h] = (bf16)mx;
    }
  }
}

// ---------------------------------------------------------------------------
// Gate softmax + floor + branch mix + per-head RMSNorm -> bf16 o_flat
// ---------------------------------------------------------------------------
__global__ __launch_bounds__(256) void combine_kernel(
    const float* __restrict__ logits, const float* __restrict__ g2_b,
    const float* __restrict__ fshort, const float* __restrict__ flong,
    const float* __restrict__ fwide, const float* __restrict__ ddelta,
    const float* __restrict__ vv, const float* __restrict__ value_bias,
    const float* __restrict__ log_temp, const float* __restrict__ floor_logit,
    const float* __restrict__ onorm, bf16* __restrict__ oflat) {
  __shared__ float wgt[4][5];
  __shared__ float red[256];
  __shared__ float msum[4];
  int row = blockIdx.x, tid = threadIdx.x;
  if (tid < 4) {
    int h = tid;
    float inv_t = expf(-log_temp[h]);
    float lg[5];
#pragma unroll
    for (int j = 0; j < 5; ++j)
      lg[j] = logits[(size_t)row * 64 + h * 5 + j] + g2_b[h * 5 + j];
    lg[4] += value_bias[h];
#pragma unroll
    for (int j = 0; j < 5; ++j) lg[j] *= inv_t;
    float mxl = lg[0];
#pragma unroll
    for (int j = 1; j < 5; ++j) mxl = fmaxf(mxl, lg[j]);
    float se = 0.0f;
#pragma unroll
    for (int j = 0; j < 5; ++j) { lg[j] = expf(lg[j] - mxl); se += lg[j]; }
    float f = 1.0f / (1.0f + expf(-floor_logit[h]));
    float scale = 1.0f - 4.0f * f;
#pragma unroll
    for (int j = 0; j < 4; ++j) wgt[h][j] = f + scale * lg[j] / se;
    wgt[h][4] = scale * lg[4] / se;
  }
  __syncthreads();
  int h = tid >> 6;
  float o[4], ss = 0.0f;
#pragma unroll
  for (int i = 0; i < 4; ++i) {
    size_t idx = (size_t)row * 1024 + tid * 4 + i;
    float val = wgt[h][0] * fshort[idx] + wgt[h][1] * flong[idx] +
                wgt[h][2] * fwide[idx] + wgt[h][3] * ddelta[idx] +
                wgt[h][4] * vv[idx];
    o[i] = val;
    ss += val * val;
  }
  red[tid] = ss;
  __syncthreads();
  if ((tid & 63) == 0) {
    float s = 0.0f;
    for (int j = 0; j < 64; ++j) s += red[tid + j];
    msum[tid >> 6] = s;
  }
  __syncthreads();
  float rn = rsqrtf(msum[h] * (1.0f / 256.0f) + 1e-5f);
#pragma unroll
  for (int i = 0; i < 4; ++i) {
    int c = tid * 4 + i;
    oflat[(size_t)row * 1024 + c] = (bf16)(o[i] * rn * onorm[c & 255]);
  }
}

// ---------------------------------------------------------------------------
// Host-side orchestration
// ---------------------------------------------------------------------------
extern "C" void kernel_launch(void* const* d_in, const int* in_sizes, int n_in,
                              void* d_out, int out_size, void* d_ws, size_t ws_size,
                              hipStream_t stream) {
  (void)in_sizes; (void)n_in; (void)out_size; (void)ws_size;
  const float* hidden      = (const float*)d_in[0];
  const float* q_w         = (const float*)d_in[1];
  const float* k_w         = (const float*)d_in[2];
  const float* v_w         = (const float*)d_in[3];
  const float* b_w         = (const float*)d_in[4];
  const float* convq       = (const float*)d_in[5];
  const float* convk       = (const float*)d_in[6];
  const float* convv       = (const float*)d_in[7];
  const float* fir_s       = (const float*)d_in[8];
  const float* fir_l       = (const float*)d_in[9];
  const float* fir_w       = (const float*)d_in[10];
  const float* g1_w        = (const float*)d_in[11];
  const float* g1_b        = (const float*)d_in[12];
  const float* g2_w        = (const float*)d_in[13];
  const float* g2_b        = (const float*)d_in[14];
  const float* value_bias  = (const float*)d_in[15];
  const float* log_temp    = (const float*)d_in[16];
  const float* floor_logit = (const float*)d_in[17];
  const float* onorm       = (const float*)d_in[18];
  const float* o_w         = (const float*)d_in[19];

  char* ws = (char*)d_ws;
  size_t off = 0;
  auto take = [&](size_t sz) { char* p = ws + off; off += sz; return p; };
  bf16* h_bf    = (bf16*)take(4096ull * 1024 * 2);
  bf16* qw_bf   = (bf16*)take(1024ull * 1024 * 2);
  bf16* kw_bf   = (bf16*)take(1024ull * 1024 * 2);
  bf16* vw_bf   = (bf16*)take(1024ull * 1024 * 2);
  bf16* ow_bf   = (bf16*)take(1024ull * 1024 * 2);
  bf16* g1w_bf  = (bf16*)take(2048ull * 1088 * 2);
  bf16* g2w_bf  = (bf16*)take(64ull * 2048 * 2);
  float* qlin   = (float*)take(4096ull * 1024 * 4);
  float* klin   = (float*)take(4096ull * 1024 * 4);
  float* vlin   = (float*)take(4096ull * 1024 * 4);
  float* qc     = (float*)take(4096ull * 1024 * 4);
  float* kc     = (float*)take(4096ull * 1024 * 4);  // contiguous after qc
  float* vc     = (float*)take(4096ull * 1024 * 4);
  float* beta   = (float*)take(4096ull * 4 * 4);
  float* ddelta = (float*)take(4096ull * 1024 * 4);
  bf16* gate_in = (bf16*)take(4096ull * 1088 * 2);
  bf16* hdn_bf  = (bf16*)take(4096ull * 2048 * 2);
  float* logits = (float*)take(4096ull * 64 * 4);
  bf16* oflat   = (bf16*)take(4096ull * 1024 * 2);
  // Buffer reuse after their producers are dead:
  float* hdn_f32 = qc;      // 32 MB spans qc+kc (both dead by gate MLP time)
  float* fshort  = qlin;    // q/k/v lin dead after convs
  float* flong   = klin;
  float* fwide   = vlin;

  const long TOT = 4096L * 1024L;  // B*L*HID elements

  // 1) bf16 staging of activations + weights (halves HBM traffic for GEMMs)
  f32_to_bf16_k<<<(int)((TOT + 255) / 256), 256, 0, stream>>>(hidden, h_bf, (int)TOT);
  f32_to_bf16_k<<<(1048576 + 255) / 256, 256, 0, stream>>>(q_w, qw_bf, 1048576);
  f32_to_bf16_k<<<(1048576 + 255) / 256, 256, 0, stream>>>(k_w, kw_bf, 1048576);
  f32_to_bf16_k<<<(1048576 + 255) / 256, 256, 0, stream>>>(v_w, vw_bf, 1048576);
  f32_to_bf16_k<<<(1048576 + 255) / 256, 256, 0, stream>>>(o_w, ow_bf, 1048576);
  pad_g1_k<<<(2048 * 1088 + 255) / 256, 256, 0, stream>>>(g1_w, g1w_bf);
  pad_g2_k<<<(64 * 2048 + 255) / 256, 256, 0, stream>>>(g2_w, g2w_bf);

  // 2) q/k/v projections (WMMA GEMMs, TDM-staged B tiles)
  dim3 gqkv(4096 / 64, 1024 / 64);
  gemm_bf16<<<gqkv, 128, 0, stream>>>(h_bf, qw_bf, qlin, 4096, 1024, 1024);
  gemm_bf16<<<gqkv, 128, 0, stream>>>(h_bf, kw_bf, klin, 4096, 1024, 1024);
  gemm_bf16<<<gqkv, 128, 0, stream>>>(h_bf, vw_bf, vlin, 4096, 1024, 1024);
  beta_k<<<(16384 + 255) / 256, 256, 0, stream>>>(hidden, b_w, beta);

  // 3) causal short conv (K=4) + silu
  int cvblk = (int)((TOT + 255) / 256);
  dwconv_causal_k<<<cvblk, 256, 0, stream>>>(qlin, convq, qc, 2048, 1024, 4, 1, TOT);
  dwconv_causal_k<<<cvblk, 256, 0, stream>>>(klin, convk, kc, 2048, 1024, 4, 1, TOT);
  dwconv_causal_k<<<cvblk, 256, 0, stream>>>(vlin, convv, vc, 2048, 1024, 4, 1, TOT);

  // 4) chunked delta rule: one block per (b,h), state S in 320KB-class LDS
  delta_kernel<<<8, 256, 0, stream>>>(qc, kc, vc, beta, ddelta);

  // 5) FIR branches on v (reuse dead q/k/v lin buffers)
  dwconv_causal_k<<<cvblk, 256, 0, stream>>>(vc, fir_s, fshort, 2048, 1024, 3, 0, TOT);
  dwconv_causal_k<<<cvblk, 256, 0, stream>>>(vc, fir_l, flong, 2048, 1024, 31, 0, TOT);
  dwconv_causal_k<<<cvblk, 256, 0, stream>>>(vc, fir_w, fwide, 2048, 1024, 64, 0, TOT);

  // 6) gate input assembly (+ per-head stats)
  stats_kernel<<<4096, 128, 0, stream>>>(fshort, flong, fwide, ddelta, vc, h_bf,
                                         gate_in);

  // 7) gate MLP (WMMA GEMMs, padded dims), gelu, logits
  gemm_bf16<<<dim3(4096 / 64, 2048 / 64), 128, 0, stream>>>(gate_in, g1w_bf, hdn_f32,
                                                            4096, 2048, 1088);
  gelu_bias_cvt_k<<<(int)((4096L * 2048 + 255) / 256), 256, 0, stream>>>(
      hdn_f32, g1_b, hdn_bf, 2048, 4096L * 2048);
  gemm_bf16<<<dim3(4096 / 64, 1), 128, 0, stream>>>(hdn_bf, g2w_bf, logits, 4096, 64,
                                                    2048);

  // 8) softmax gate + floor + mix + RMSNorm
  combine_kernel<<<4096, 256, 0, stream>>>(logits, g2_b, fshort, flong, fwide, ddelta,
                                           vc, value_bias, log_temp, floor_logit,
                                           onorm, oflat);

  // 9) output projection -> d_out (f32)
  gemm_bf16<<<dim3(4096 / 64, 1024 / 64), 128, 0, stream>>>(oflat, ow_bf,
                                                            (float*)d_out, 4096, 1024,
                                                            1024);
}
